// QwenWindowAttention_14697378086945
// MI455X (gfx1250) — matrix operand order repaired and checked by reading, not compiled, not run on hardware
//
#include <hip/hip_runtime.h>
#include <hip/hip_bf16.h>

typedef __attribute__((ext_vector_type(16))) __bf16 v16bf;
typedef __attribute__((ext_vector_type(8)))  float  v8f;
typedef __attribute__((ext_vector_type(4)))  int    v4i;
typedef __attribute__((ext_vector_type(4)))  unsigned int u32x4;
typedef __attribute__((ext_vector_type(8)))  int    i32x8;
typedef __attribute__((ext_vector_type(4)))  int    i32x4;

union Frag { v4i i[2]; v16bf v; };
union PkBf { __bf16 h[2]; unsigned int u; };

#define S_TOK 16384
#define C_DIM 1280
#define HD    80
#define NH    16

static __device__ __forceinline__ v8f v8zero() {
  v8f z = {0.f,0.f,0.f,0.f,0.f,0.f,0.f,0.f};
  return z;
}

static __device__ __forceinline__ v8f wmma_bf16(v16bf a, v16bf b, v8f c) {
  // D = A(16x32 bf16) x B(32x16 bf16) + C(16x16 f32)
  return __builtin_amdgcn_wmma_f32_16x16x32_bf16(
      /*neg_a=*/false, a, /*neg_b=*/false, b,
      /*c_mod=*/(short)0, c, /*reuse_a=*/false, /*reuse_b=*/false);
}

// ---------------------------------------------------------------------------
// TDM: issue a tensor_load_to_lds for a (up to 3-D) bf16 tile.
// Descriptor packing per CDNA5 ISA ch.8 (D# groups 0..3), data_size = 2 bytes.
// Whole-wave op (EXEC ignored); tracked by TENSORcnt (in-order per wave).
// ---------------------------------------------------------------------------
static __device__ __forceinline__ void tdm_load_bf16(
    unsigned int lds_off, const void* gaddr,
    unsigned int tile0, unsigned int tile1, unsigned int tile2,
    unsigned int tdim0, unsigned int tdim1, unsigned int tdim2,
    unsigned long long stride0, unsigned long long stride1) {
  unsigned long long ga = (unsigned long long)(size_t)gaddr;
  u32x4 g0;
  g0[0] = 1u;                                        // count=1, user mode
  g0[1] = lds_off;                                   // lds_addr (bytes)
  g0[2] = (unsigned int)(ga & 0xffffffffull);        // global_addr[31:0]
  g0[3] = (unsigned int)((ga >> 32) & 0x1ffffffull)  // global_addr[56:32]
          | (2u << 30);                              // type = 2 ("image")
  i32x8 g1;
  g1[0] = (int)(1u << 16);                           // data_size=1 (2 bytes)
  g1[1] = (int)((tdim0 & 0xffffu) << 16);            // tensor_dim0[15:0] @ bit48
  g1[2] = (int)(((tdim0 >> 16) & 0xffffu) | ((tdim1 & 0xffffu) << 16));
  g1[3] = (int)(((tdim1 >> 16) & 0xffffu) | ((tile0 & 0xffffu) << 16));
  g1[4] = (int)((tile1 & 0xffffu) | ((tile2 & 0xffffu) << 16));
  g1[5] = (int)(stride0 & 0xffffffffull);            // tensor_dim0_stride[31:0]
  g1[6] = (int)(((stride0 >> 32) & 0xffffull)
          | (unsigned int)((stride1 & 0xffffull) << 16));
  g1[7] = (int)((stride1 >> 16) & 0xffffffffull);
  i32x4 g2;
  g2[0] = (int)tdim2;                                // tensor_dim2
  g2[1] = 0;                                         // tensor_dim3
  g2[2] = 0;                                         // tensor_dim2_stride lo
  g2[3] = 0;                                         // stride hi | tile_dim3=0
  i32x4 g3 = {0, 0, 0, 0};
#if __clang_major__ >= 23
  i32x8 gx = {0, 0, 0, 0, 0, 0, 0, 0};
  __builtin_amdgcn_tensor_load_to_lds(g0, g1, g2, g3, gx, 0);
#else
  __builtin_amdgcn_tensor_load_to_lds(g0, g1, g2, g3, 0);
#endif
}

static __device__ __forceinline__ unsigned int lds_offset_of(const void* p) {
  return (unsigned int)(size_t)p;   // AS(3) pointer value == byte offset in LDS
}

// ---------------------------------------------------------------------------
// Kernel 0: convert fp32 weight [K][N] -> bf16 transposed [N][K]
// ---------------------------------------------------------------------------
__global__ __launch_bounds__(256) void convert_wt_kernel(
    const float* __restrict__ W, __bf16* __restrict__ Wt, int K, int N) {
  size_t idx = (size_t)blockIdx.x * blockDim.x + threadIdx.x;
  size_t total = (size_t)K * N;
  if (idx >= total) return;
  int k = (int)(idx / N);
  int n = (int)(idx % N);
  Wt[(size_t)n * K + k] = (__bf16)W[idx];
}

// ---------------------------------------------------------------------------
// Kernel 1: gather(x by window_index) -> QKV GEMM (bf16 WMMA) -> RoPE
// BM=128, BN=64 (x3 matrices), BK=32. 256 threads = 8 waves (4x2).
// Double-buffered pipeline: B tiles prefetched by one 3-D TDM descriptor per
// step into the alternate buffer; A slice prefetched into registers.
// ---------------------------------------------------------------------------
__global__ __launch_bounds__(256) void qkv_rope_kernel(
    const float* __restrict__ x,      // [S, C]
    const float* __restrict__ cosp,   // [S, 80]
    const float* __restrict__ sinp,   // [S, 80]
    const __bf16* __restrict__ Wt,    // [3C][C]  (Wt[n][k] = W_qkv[k][n])
    const int*   __restrict__ wi,     // [S]
    __bf16* __restrict__ qro,
    __bf16* __restrict__ kro,
    __bf16* __restrict__ vo) {
  __shared__ __align__(16) __bf16 As[2][128][32];
  __shared__ __align__(16) __bf16 Bqkv[2][3][64][32];   // TDM dest: [mat][n][k]
  __shared__ int wiRow[128];

  const int tid = threadIdx.x;
  const int m0 = blockIdx.x * 128;
  const int n0 = blockIdx.y * 64;
  if (tid < 128) wiRow[tid] = wi[m0 + tid];
  __syncthreads();

  const int lane = tid & 31;
  const int wv = tid >> 5;
  const int wM = wv >> 1;     // 0..3 -> 32-row strip
  const int wN = wv & 1;      // 0..1 -> 32-col strip
  const int lh = lane & 15;
  const int hh = lane >> 4;

  v8f accQ[2][2], accK[2][2], accV[2][2];
#pragma unroll
  for (int i = 0; i < 2; i++)
#pragma unroll
    for (int j = 0; j < 2; j++) {
      accQ[i][j] = v8zero(); accK[i][j] = v8zero(); accV[i][j] = v8zero();
    }

  const int rT = tid >> 3;          // 0..31
  const int c4 = (tid & 7) * 4;     // 0..28
  const unsigned int bOff0 = lds_offset_of(&Bqkv[0][0][0][0]);
  const unsigned int bOff1 = lds_offset_of(&Bqkv[1][0][0][0]);

  // Row bases for this thread's 4 gathered A rows (fixed across k-steps).
  const float* aBase[4];
  int rows[4];
#pragma unroll
  for (int s = 0; s < 4; s++) {
    rows[s] = rT + s * 32;
    aBase[s] = x + (size_t)wiRow[rows[s]] * C_DIM + c4;
  }

  // ---- Prologue: TDM for buffer 0, A slice 0 into registers
  if (wv == 0) {
    tdm_load_bf16(bOff0, Wt + (size_t)n0 * C_DIM,
                  32u, 64u, 3u, (unsigned)C_DIM, 3u * C_DIM, 3u,
                  (unsigned long long)C_DIM,
                  (unsigned long long)C_DIM * C_DIM);
  }
  float4 aReg[4];
#pragma unroll
  for (int s = 0; s < 4; s++) aReg[s] = *(const float4*)(aBase[s]);

  const int NSTEP = C_DIM / 32;     // 40
  for (int st = 0; st < NSTEP; st++) {
    const int buf = st & 1;
    const int k0 = st * 32;

    // Stage A registers -> LDS (packed bf16 pairs)
#pragma unroll
    for (int s = 0; s < 4; s++) {
      PkBf p0, p1;
      p0.h[0] = (__bf16)aReg[s].x; p0.h[1] = (__bf16)aReg[s].y;
      p1.h[0] = (__bf16)aReg[s].z; p1.h[1] = (__bf16)aReg[s].w;
      *(unsigned int*)&As[buf][rows[s]][c4]     = p0.u;
      *(unsigned int*)&As[buf][rows[s]][c4 + 2] = p1.u;
    }

    if (st + 1 < NSTEP) {
      // Prefetch next step: TDM into the other buffer + A regs.
      if (wv == 0) {
        tdm_load_bf16(buf ? bOff0 : bOff1,
                      Wt + (size_t)n0 * C_DIM + k0 + 32,
                      32u, 64u, 3u, (unsigned)C_DIM, 3u * C_DIM, 3u,
                      (unsigned long long)C_DIM,
                      (unsigned long long)C_DIM * C_DIM);
      }
#pragma unroll
      for (int s = 0; s < 4; s++)
        aReg[s] = *(const float4*)(aBase[s] + k0 + 32);
      // In-order TDM completion: <=1 outstanding => current buffer resident.
      if (wv == 0) __builtin_amdgcn_s_wait_tensorcnt(1);
    } else {
      if (wv == 0) __builtin_amdgcn_s_wait_tensorcnt(0);
    }
    __syncthreads();

    Frag a[2], bq[2], bk[2], bv[2];
#pragma unroll
    for (int tm = 0; tm < 2; tm++) {
      const int m = wM * 32 + tm * 16 + lh;
      a[tm].i[0] = *(const v4i*)&As[buf][m][hh * 16];        // K = 16*hh + 0..7
      a[tm].i[1] = *(const v4i*)&As[buf][m][hh * 16 + 8];    // K = 16*hh + 8..15
    }
#pragma unroll
    for (int tn = 0; tn < 2; tn++) {
      const int n = wN * 32 + tn * 16 + lh;
      const v4i* pq = (const v4i*)&Bqkv[buf][0][n][hh * 16];
      const v4i* pk = (const v4i*)&Bqkv[buf][1][n][hh * 16];
      const v4i* pv = (const v4i*)&Bqkv[buf][2][n][hh * 16];
      bq[tn].i[0] = pq[0]; bq[tn].i[1] = pq[1];
      bk[tn].i[0] = pk[0]; bk[tn].i[1] = pk[1];
      bv[tn].i[0] = pv[0]; bv[tn].i[1] = pv[1];
    }
#pragma unroll
    for (int tm = 0; tm < 2; tm++)
#pragma unroll
      for (int tn = 0; tn < 2; tn++) {
        accQ[tm][tn] = wmma_bf16(a[tm].v, bq[tn].v, accQ[tm][tn]);
        accK[tm][tn] = wmma_bf16(a[tm].v, bk[tn].v, accK[tm][tn]);
        accV[tm][tn] = wmma_bf16(a[tm].v, bv[tn].v, accV[tm][tn]);
      }
    __syncthreads();   // readers done before buffer is re-filled next step
  }

  // Epilogue: RoPE and bf16 stores (C layout: elem r -> m = r + 8*hh, n = lh)
#pragma unroll
  for (int tm = 0; tm < 2; tm++) {
#pragma unroll
    for (int tn = 0; tn < 2; tn++) {
      const int c = n0 + wN * 32 + tn * 16 + lh;
      const int d = c % HD;
#pragma unroll
      for (int r = 0; r < 8; r++) {
        const int mi = wM * 32 + tm * 16 + hh * 8 + r;
        const int row = m0 + mi;
        const int w = wiRow[mi];
        const float cv = cosp[(size_t)w * HD + d];
        const float sv = sinp[(size_t)w * HD + d];
        const float qf = accQ[tm][tn][r];
        const float kf = accK[tm][tn][r];
        const size_t o = (size_t)row * C_DIM + c;
        qro[o] = (__bf16)(qf * cv + kf * sv);
        kro[o] = (__bf16)(kf * cv - qf * sv);
        vo[o]  = (__bf16)(accV[tm][tn][r]);
      }
    }
  }
}

// ---------------------------------------------------------------------------
// Kernel 2: windowed attention. One wave per (block of 16 tokens, head).
// ---------------------------------------------------------------------------
__global__ __launch_bounds__(128) void attn_kernel(
    const __bf16* __restrict__ qr, const __bf16* __restrict__ kr,
    const __bf16* __restrict__ vv, const int* __restrict__ wi,
    __bf16* __restrict__ y) {
  __shared__ __align__(16) __bf16 Qs[4][16][96];
  __shared__ __align__(16) __bf16 Ks[4][16][96];
  __shared__ __align__(16) __bf16 VT[4][80][16];   // [d][token]
  __shared__ __align__(16) __bf16 Ps[4][16][16];

  const int wv = threadIdx.x >> 5;
  const int lane = threadIdx.x & 31;
  const int pair = blockIdx.x * 4 + wv;
  const int blk = pair >> 4;
  const int head = pair & 15;
  const int t0 = blk * 16;
  const size_t base = (size_t)t0 * C_DIM + head * HD;

  for (int e = lane; e < 16 * HD; e += 32) {
    const int tok = e / HD;
    const int d = e % HD;
    const size_t g = base + (size_t)tok * C_DIM + d;
    Qs[wv][tok][d] = qr[g];
    Ks[wv][tok][d] = kr[g];
    VT[wv][d][tok] = vv[g];
  }
  // zero pad d = 80..95 (avoid NaN from poisoned LDS in padded K-range)
  for (int e = lane; e < 16 * 16; e += 32) {
    const int tok = e >> 4;
    const int d = 80 + (e & 15);
    Qs[wv][tok][d] = (__bf16)0.f;
    Ks[wv][tok][d] = (__bf16)0.f;
  }
  __syncthreads();

  const int lh = lane & 15;
  const int hh = lane >> 4;

  v8f sc = v8zero();
#pragma unroll
  for (int ks = 0; ks < 96; ks += 32) {
    Frag fa, fb;
    fa.i[0] = *(const v4i*)&Qs[wv][lh][ks + hh * 16];
    fa.i[1] = *(const v4i*)&Qs[wv][lh][ks + hh * 16 + 8];
    const v4i* kp = (const v4i*)&Ks[wv][lh][ks + hh * 16];
    fb.i[0] = kp[0];
    fb.i[1] = kp[1];
    sc = wmma_bf16(fa.v, fb.v, sc);
  }

  // softmax along columns: row m = r + 8*hh lives in one VGPR across a
  // 16-lane half-wave -> width-16 xor-shuffle reductions.
  const float scale = 0.11180339887498949f;  // 1/sqrt(80)
#pragma unroll
  for (int r = 0; r < 8; r++) {
    float s = sc[r] * scale;
    float mx = s;
#pragma unroll
    for (int off = 8; off > 0; off >>= 1) mx = fmaxf(mx, __shfl_xor(mx, off, 16));
    const float e = __expf(s - mx);
    float sum = e;
#pragma unroll
    for (int off = 8; off > 0; off >>= 1) sum += __shfl_xor(sum, off, 16);
    Ps[wv][hh * 8 + r][lh] = (__bf16)(e / sum);
  }
  __syncthreads();

  // P as A-fragment (M=16, K=32 with K=16..31 zero-padded)
  Frag fp;
  v4i z4 = {0, 0, 0, 0};
  fp.i[0] = *(const v4i*)&Ps[wv][lh][hh * 8];
  fp.i[1] = z4;

  int orow[8];
#pragma unroll
  for (int r = 0; r < 8; r++) orow[r] = wi[t0 + hh * 8 + r];

#pragma unroll
  for (int nt = 0; nt < 5; nt++) {
    Frag fv;
    fv.i[0] = *(const v4i*)&VT[wv][nt * 16 + lh][hh * 8];
    fv.i[1] = z4;
    v8f ya = wmma_bf16(fp.v, fv.v, v8zero());
    const int d = nt * 16 + lh;
#pragma unroll
    for (int r = 0; r < 8; r++) {
      y[(size_t)orow[r] * C_DIM + head * HD + d] = (__bf16)ya[r];
    }
  }
}

// ---------------------------------------------------------------------------
// Kernel 3: out = y (bf16 [S][C]) @ W_proj -> fp32.
// Fully TDM-staged, double-buffered pipeline (2 descriptors in flight).
// ---------------------------------------------------------------------------
__global__ __launch_bounds__(256) void proj_kernel(
    const __bf16* __restrict__ y, const __bf16* __restrict__ Wt,
    float* __restrict__ out) {
  __shared__ __align__(16) __bf16 As[2][128][32];
  __shared__ __align__(16) __bf16 Bs[2][64][32];

  const int tid = threadIdx.x;
  const int m0 = blockIdx.x * 128;
  const int n0 = blockIdx.y * 64;
  const int lane = tid & 31;
  const int wv = tid >> 5;
  const int wM = wv >> 1;
  const int wN = wv & 1;
  const int lh = lane & 15;
  const int hh = lane >> 4;

  v8f acc[2][2];
#pragma unroll
  for (int i = 0; i < 2; i++)
#pragma unroll
    for (int j = 0; j < 2; j++) acc[i][j] = v8zero();

  const unsigned int aOff[2] = { lds_offset_of(&As[0][0][0]),
                                 lds_offset_of(&As[1][0][0]) };
  const unsigned int bOff[2] = { lds_offset_of(&Bs[0][0][0]),
                                 lds_offset_of(&Bs[1][0][0]) };

  if (wv == 0) {
    tdm_load_bf16(aOff[0], y + (size_t)m0 * C_DIM,
                  32u, 128u, 0u, (unsigned)C_DIM, (unsigned)S_TOK, 0u,
                  (unsigned long long)C_DIM, 0ull);
    tdm_load_bf16(bOff[0], Wt + (size_t)n0 * C_DIM,
                  32u, 64u, 0u, (unsigned)C_DIM, (unsigned)C_DIM, 0u,
                  (unsigned long long)C_DIM, 0ull);
  }

  const int NSTEP = C_DIM / 32;
  for (int st = 0; st < NSTEP; st++) {
    const int buf = st & 1;
    const int k0 = st * 32;
    if (st + 1 < NSTEP) {
      if (wv == 0) {
        tdm_load_bf16(aOff[buf ^ 1], y + (size_t)m0 * C_DIM + k0 + 32,
                      32u, 128u, 0u, (unsigned)C_DIM, (unsigned)S_TOK, 0u,
                      (unsigned long long)C_DIM, 0ull);
        tdm_load_bf16(bOff[buf ^ 1], Wt + (size_t)n0 * C_DIM + k0 + 32,
                      32u, 64u, 0u, (unsigned)C_DIM, (unsigned)C_DIM, 0u,
                      (unsigned long long)C_DIM, 0ull);
        // 2 newer in flight; <=2 outstanding => current pair resident.
        __builtin_amdgcn_s_wait_tensorcnt(2);
      }
    } else {
      if (wv == 0) __builtin_amdgcn_s_wait_tensorcnt(0);
    }
    __syncthreads();

    Frag a[2], b[2];
#pragma unroll
    for (int tm = 0; tm < 2; tm++) {
      const int m = wM * 32 + tm * 16 + lh;
      a[tm].i[0] = *(const v4i*)&As[buf][m][hh * 16];
      a[tm].i[1] = *(const v4i*)&As[buf][m][hh * 16 + 8];
    }
#pragma unroll
    for (int tn = 0; tn < 2; tn++) {
      const int n = wN * 32 + tn * 16 + lh;
      const v4i* p = (const v4i*)&Bs[buf][n][hh * 16];
      b[tn].i[0] = p[0];
      b[tn].i[1] = p[1];
    }
#pragma unroll
    for (int tm = 0; tm < 2; tm++)
#pragma unroll
      for (int tn = 0; tn < 2; tn++)
        acc[tm][tn] = wmma_bf16(a[tm].v, b[tn].v, acc[tm][tn]);
    __syncthreads();   // readers done before buffer is re-filled next step
  }

#pragma unroll
  for (int tm = 0; tm < 2; tm++) {
#pragma unroll
    for (int tn = 0; tn < 2; tn++) {
      const int c = n0 + wN * 32 + tn * 16 + lh;
#pragma unroll
      for (int r = 0; r < 8; r++) {
        const int row = m0 + wM * 32 + tm * 16 + hh * 8 + r;
        out[(size_t)row * C_DIM + c] = acc[tm][tn][r];
      }
    }
  }
}

// ---------------------------------------------------------------------------
extern "C" void kernel_launch(void* const* d_in, const int* in_sizes, int n_in,
                              void* d_out, int out_size, void* d_ws, size_t ws_size,
                              hipStream_t stream) {
  const float* x     = (const float*)d_in[0];
  const float* cosp  = (const float*)d_in[1];
  const float* sinp  = (const float*)d_in[2];
  const float* Wqkv  = (const float*)d_in[3];
  const float* Wproj = (const float*)d_in[4];
  const int*   wi    = (const int*)d_in[5];
  float* out = (float*)d_out;

  char* ws = (char*)d_ws;
  // ws layout (bytes):
  //   WqkvT  : 3840*1280*2 =  9,830,400   @ 0
  //   WprojT : 1280*1280*2 =  3,276,800   @ 9,830,400
  //   qr     : 16384*1280*2 = 41,943,040  @ 13,107,200
  //   kr     :                             @ 55,050,240
  //   v      :                             @ 96,993,280
  //   y      :                             @ 138,936,320   (total ~172.5 MB)
  __bf16* WqkvT  = (__bf16*)(ws);
  __bf16* WprojT = (__bf16*)(ws + 9830400u);
  __bf16* qr     = (__bf16*)(ws + 13107200u);
  __bf16* kr     = (__bf16*)(ws + 55050240u);
  __bf16* vv     = (__bf16*)(ws + 96993280u);
  __bf16* yb     = (__bf16*)(ws + 138936320u);

  {
    const size_t tot = (size_t)C_DIM * 3 * C_DIM;
    convert_wt_kernel<<<(unsigned)((tot + 255) / 256), 256, 0, stream>>>(
        Wqkv, WqkvT, C_DIM, 3 * C_DIM);
  }
  {
    const size_t tot = (size_t)C_DIM * C_DIM;
    convert_wt_kernel<<<(unsigned)((tot + 255) / 256), 256, 0, stream>>>(
        Wproj, WprojT, C_DIM, C_DIM);
  }

  qkv_rope_kernel<<<dim3(S_TOK / 128, C_DIM / 64), 256, 0, stream>>>(
      x, cosp, sinp, WqkvT, wi, qr, kr, vv);

  attn_kernel<<<(S_TOK / 16) * NH / 4, 128, 0, stream>>>(qr, kr, vv, wi, yb);

  proj_kernel<<<dim3(S_TOK / 128, C_DIM / 64), 256, 0, stream>>>(yb, WprojT, out);
}
